// MinimalDroneModel_79783312490952
// MI455X (gfx1250) — matrix operand order repaired and checked by reading, not compiled, not run on hardware
//
#include <hip/hip_runtime.h>
#include <hip/hip_bf16.h>

typedef _Float16 half_t;
typedef _Float16 half16 __attribute__((ext_vector_type(16)));
typedef _Float16 half8v __attribute__((ext_vector_type(8)));
typedef float    f32x8  __attribute__((ext_vector_type(8)));
typedef int      i32x4v __attribute__((ext_vector_type(4)));

#define GAS __attribute__((address_space(1)))
#define LAS __attribute__((address_space(3)))

#define BB 8
#define NN 1024
#define DD 128
#define TOK (BB * NN)          // 8192 tokens
#define FLT_MIN_REF (-3.4028234663852886e38f)

#define HAS_ASYNC_LDS __has_builtin(__builtin_amdgcn_global_load_async_to_lds_b128)

// ---------------------------------------------------------------------------
// Geometry branch: (TOK,3) -> relu(3x64) -> (64x32) -> LayerNorm -> geom (TOK,32)
// ---------------------------------------------------------------------------
__global__ void geom_kernel(const float* __restrict__ rays,
                            const float* __restrict__ w1, const float* __restrict__ b1,
                            const float* __restrict__ w2, const float* __restrict__ b2,
                            const float* __restrict__ lng, const float* __restrict__ lnb,
                            float* __restrict__ geom) {
  int t = blockIdx.x * blockDim.x + threadIdx.x;
  if (t >= TOK) return;
  float r0 = rays[t * 3 + 0], r1 = rays[t * 3 + 1], r2 = rays[t * 3 + 2];
  float h[64];
#pragma unroll
  for (int j = 0; j < 64; ++j)
    h[j] = fmaxf(r0 * w1[j] + r1 * w1[64 + j] + r2 * w1[128 + j] + b1[j], 0.f);
  float o[32];
  float mu = 0.f;
#pragma unroll
  for (int c = 0; c < 32; ++c) {
    float s = b2[c];
    for (int j = 0; j < 64; ++j) s += h[j] * w2[j * 32 + c];
    o[c] = s; mu += s;
  }
  mu *= (1.f / 32.f);
  float var = 0.f;
#pragma unroll
  for (int c = 0; c < 32; ++c) { float d = o[c] - mu; var += d * d; }
  var *= (1.f / 32.f);
  float inv = rsqrtf(var + 1e-5f);
#pragma unroll
  for (int c = 0; c < 32; ++c)
    geom[t * 32 + c] = (o[c] - mu) * inv * lng[c] + lnb[c];
}

// ---------------------------------------------------------------------------
// Conv backbone (stride-2, pad-1 3x3 convs). Naive direct conv; tiny FLOPs.
// ---------------------------------------------------------------------------
__global__ void conv1_kernel(const float* __restrict__ img, const float* __restrict__ w,
                             const float* __restrict__ bias, float* __restrict__ out) {
  int idx = blockIdx.x * blockDim.x + threadIdx.x;
  const int TOTAL = BB * 16 * 112 * 112;
  if (idx >= TOTAL) return;
  int ox = idx % 112; int t = idx / 112;
  int oy = t % 112; t /= 112;
  int oc = t % 16; int b = t / 16;
  float s = bias[oc];
  for (int ic = 0; ic < 3; ++ic)
    for (int ky = 0; ky < 3; ++ky) {
      int iy = oy * 2 - 1 + ky; if (iy < 0 || iy >= 224) continue;
      for (int kx = 0; kx < 3; ++kx) {
        int ix = ox * 2 - 1 + kx; if (ix < 0 || ix >= 224) continue;
        s += img[((b * 3 + ic) * 224 + iy) * 224 + ix] * w[((oc * 3 + ic) * 3 + ky) * 3 + kx];
      }
    }
  out[idx] = fmaxf(s, 0.f);
}

__global__ void conv2_kernel(const float* __restrict__ f1, const float* __restrict__ w,
                             const float* __restrict__ bias, float* __restrict__ out) {
  int idx = blockIdx.x * blockDim.x + threadIdx.x;
  const int TOTAL = BB * 32 * 56 * 56;
  if (idx >= TOTAL) return;
  int ox = idx % 56; int t = idx / 56;
  int oy = t % 56; t /= 56;
  int oc = t % 32; int b = t / 32;
  float s = bias[oc];
  for (int ic = 0; ic < 16; ++ic)
    for (int ky = 0; ky < 3; ++ky) {
      int iy = oy * 2 - 1 + ky; if (iy < 0 || iy >= 112) continue;
      for (int kx = 0; kx < 3; ++kx) {
        int ix = ox * 2 - 1 + kx; if (ix < 0 || ix >= 112) continue;
        s += f1[((b * 16 + ic) * 112 + iy) * 112 + ix] * w[((oc * 16 + ic) * 3 + ky) * 3 + kx];
      }
    }
  out[idx] = fmaxf(s, 0.f);
}

// conv3 fused with global mean: one block per (b, oc)
__global__ void conv3_mean_kernel(const float* __restrict__ f2, const float* __restrict__ w,
                                  const float* __restrict__ bias, float* __restrict__ appear) {
  int oc = blockIdx.x % 96;
  int b  = blockIdx.x / 96;
  float s = 0.f;
  for (int p = threadIdx.x; p < 28 * 28; p += blockDim.x) {
    int oy = p / 28, ox = p % 28;
    float v = 0.f;
    for (int ic = 0; ic < 32; ++ic)
      for (int ky = 0; ky < 3; ++ky) {
        int iy = oy * 2 - 1 + ky; if (iy < 0 || iy >= 56) continue;
        for (int kx = 0; kx < 3; ++kx) {
          int ix = ox * 2 - 1 + kx; if (ix < 0 || ix >= 56) continue;
          v += f2[((b * 32 + ic) * 56 + iy) * 56 + ix] * w[((oc * 32 + ic) * 3 + ky) * 3 + kx];
        }
      }
    s += v;
  }
  __shared__ float red[256];
  red[threadIdx.x] = s; __syncthreads();
  for (int st = 128; st > 0; st >>= 1) {
    if ((int)threadIdx.x < st) red[threadIdx.x] += red[threadIdx.x + st];
    __syncthreads();
  }
  if (threadIdx.x == 0) appear[b * 96 + oc] = red[0] * (1.f / 784.f) + bias[oc];
}

// ---------------------------------------------------------------------------
// joint_feat = concat(geom, appear) -> f32 + f16 copy
// ---------------------------------------------------------------------------
__global__ void joint_kernel(const float* __restrict__ geom, const float* __restrict__ appear,
                             float* __restrict__ joint, half_t* __restrict__ jointH) {
  int idx = blockIdx.x * blockDim.x + threadIdx.x;
  if (idx >= TOK * DD) return;
  int c = idx & 127; int t = idx >> 7; int b = t >> 10;
  float v = (c < 32) ? geom[t * 32 + c] : appear[b * 96 + (c - 32)];
  joint[idx] = v; jointH[idx] = (half_t)v;
}

// ---------------------------------------------------------------------------
// Pack a 128x128 f32 weight (K-major) into WMMA B-fragment layout:
//   frag f = (k/32)*tilesN + n/16 ; within frag: lane*16 + (k%8) + 8*((k%32)>=16),
//   lane = (n%16) + 16*((k%16)>=8).  Each lane's 16 halfs are contiguous
//   (32B) so the GEMM B-load is two global_load_b128.
// ---------------------------------------------------------------------------
__device__ __forceinline__ size_t pack_idx(int k, int n, int tilesN) {
  int kb = k >> 5, kin = k & 31, tn = n >> 4;
  int lane = (n & 15) + ((kin & 8) ? 16 : 0);
  int i = (kin & 7) | ((kin >> 1) & 8);
  return ((size_t)(kb * tilesN + tn) * 32 + lane) * 16 + i;
}

__global__ void packw_kernel(const float* __restrict__ vw, const float* __restrict__ ow,
                             const float* __restrict__ dw,
                             half_t* __restrict__ vp, half_t* __restrict__ op,
                             half_t* __restrict__ dp) {
  int idx = blockIdx.x * blockDim.x + threadIdx.x;
  if (idx >= DD * DD) return;
  int k = idx >> 7, n = idx & 127;
  size_t d = pack_idx(k, n, 8);
  vp[d] = (half_t)vw[idx];
  op[d] = (half_t)ow[idx];
  dp[d] = (half_t)dw[idx];
}

// ---------------------------------------------------------------------------
// WMMA GEMM: Y(M x 128) = X_f16(M x 128) @ Wpack + bias (+resid)
// One 16x16 tile per wave; 4 x v_wmma_f32_16x16x32_f16 over K=128.
// ---------------------------------------------------------------------------
__global__ void gemm_wmma_128(const half_t* __restrict__ X, const half_t* __restrict__ Wp,
                              const float* __restrict__ bias, const float* __restrict__ resid,
                              float* __restrict__ outF, half_t* __restrict__ outH,
                              half_t* __restrict__ outVpack) {
  const int lane = threadIdx.x & 31;
  const int wave = threadIdx.x >> 5;
  const int tile = blockIdx.x * 4 + wave;
  const int tm = tile >> 3;          // 8 N-tiles
  const int tn = tile & 7;
  const int m0 = tm * 16, n0 = tn * 16;

  const int mA    = m0 + (lane & 15);
  const int kAoff = (lane < 16) ? 0 : 8;   // A: K halves {0..7,16..23} / {8..15,24..31}

  f32x8 acc = {};
#pragma unroll
  for (int kb = 0; kb < DD; kb += 32) {
    half16 a;
    const half_t* ap = X + (size_t)mA * DD + kb + kAoff;
    half8v alo = *(const half8v*)(ap);
    half8v ahi = *(const half8v*)(ap + 16);
#pragma unroll
    for (int i = 0; i < 8; ++i) { a[i] = alo[i]; a[8 + i] = ahi[i]; }
    // packed B: one contiguous 32B chunk per lane
    const half_t* bp = Wp + ((size_t)((kb >> 5) * 8 + tn) * 32 + lane) * 16;
    half16 b = *(const half16*)bp;
    acc = __builtin_amdgcn_wmma_f32_16x16x32_f16(false, a, false, b,
                                                 (short)0, acc, false, false);
  }
  const int col = n0 + (lane & 15);
#pragma unroll
  for (int r = 0; r < 8; ++r) {
    int row = m0 + ((lane < 16) ? r : (8 + r));
    float y = acc[r] + bias[col];
    if (resid) y += resid[(size_t)row * DD + col];
    if (outF) outF[(size_t)row * DD + col] = y;
    if (outH) outH[(size_t)row * DD + col] = (half_t)y;
    if (outVpack) {
      // scatter straight into attn@V B-fragment layout: per (b,h), K=j, N=dh
      int batch = row >> 10, j = row & 1023;
      int h = col >> 6, n = col & 63;
      outVpack[(size_t)(batch * 2 + h) * (32 * 4 * 512) + pack_idx(j, n, 4)] = (half_t)y;
    }
  }
}

// ---------------------------------------------------------------------------
// Fused attention logits + mask + softmax.  One block per (batch, query i).
// rays/coords for the batch staged in LDS via async-to-LDS (ASYNCcnt path).
// Row kept in LDS; attention matrix written once, as f16.
// ---------------------------------------------------------------------------
__device__ float block_max(float v, float* red) {
  int tid = threadIdx.x;
  red[tid] = v; __syncthreads();
  for (int s = 128; s > 0; s >>= 1) {
    if (tid < s) red[tid] = fmaxf(red[tid], red[tid + s]);
    __syncthreads();
  }
  float r = red[0]; __syncthreads();
  return r;
}
__device__ float block_sum(float v, float* red) {
  int tid = threadIdx.x;
  red[tid] = v; __syncthreads();
  for (int s = 128; s > 0; s >>= 1) {
    if (tid < s) red[tid] += red[tid + s];
    __syncthreads();
  }
  float r = red[0]; __syncthreads();
  return r;
}

__global__ void attn_softmax_kernel(const float* __restrict__ rays,
                                    const float* __restrict__ coords,
                                    const float* __restrict__ w1, const float* __restrict__ b1,
                                    const float* __restrict__ w2, const float* __restrict__ b2,
                                    half_t* __restrict__ attn) {
  const int i = blockIdx.x & (NN - 1);
  const int batch = blockIdx.x >> 10;
  __shared__ float sw1[96], sb1[32], sw2[64], sb2[2];
  __shared__ float srays[NN * 3];    // 12 KB
  __shared__ float scoords[NN * 3];  // 12 KB
  __shared__ float row0[NN], row1[NN];
  __shared__ float red[256];
  const int tid = threadIdx.x;
  if (tid < 96) sw1[tid] = w1[tid];
  if (tid < 32) sb1[tid] = b1[tid];
  if (tid < 64) sw2[tid] = w2[tid];
  if (tid < 2)  sb2[tid] = b2[tid];

  const size_t base = (size_t)batch * NN;
#if HAS_ASYNC_LDS
  {
    const char* gr = (const char*)(rays + base * 3);
    const char* gc = (const char*)(coords + base * 3);
    for (int c = tid; c < (NN * 3 * 4) / 16; c += 256) {   // 768 x 16B chunks
      __builtin_amdgcn_global_load_async_to_lds_b128(
          (GAS i32x4v*)(gr + c * 16),
          (LAS i32x4v*)((char*)srays + c * 16), 0, 0);
      __builtin_amdgcn_global_load_async_to_lds_b128(
          (GAS i32x4v*)(gc + c * 16),
          (LAS i32x4v*)((char*)scoords + c * 16), 0, 0);
    }
  }
#if __has_builtin(__builtin_amdgcn_s_wait_asynccnt)
  __builtin_amdgcn_s_wait_asynccnt(0);
#else
  asm volatile("s_wait_asynccnt 0" ::: "memory");
#endif
#else
  for (int e = tid; e < NN * 3; e += 256) {
    srays[e] = rays[base * 3 + e];
    scoords[e] = coords[base * 3 + e];
  }
#endif
  __syncthreads();

  float rx = srays[i * 3 + 0], ry = srays[i * 3 + 1], rz = srays[i * 3 + 2];
  float cx = scoords[i * 3 + 0], cy = scoords[i * 3 + 1], cz = scoords[i * 3 + 2];

  for (int j = tid; j < NN; j += 256) {
    float dx = cx - scoords[j * 3 + 0];
    float dy = cy - scoords[j * 3 + 1];
    float dz = cz - scoords[j * 3 + 2];
    bool nb = (dx * dx + dy * dy + dz * dz) < 0.36f;   // dist < 0.6
    float ex = rx - srays[j * 3 + 0];
    float ey = ry - srays[j * 3 + 1];
    float ez = rz - srays[j * 3 + 2];
    float nrm = sqrtf(ex * ex + ey * ey + ez * ez);
    float inv = 1.f / fmaxf(nrm, 1e-12f);
    ex *= inv; ey *= inv; ez *= inv;
    float l0 = sb2[0], l1 = sb2[1];
#pragma unroll
    for (int k = 0; k < 32; ++k) {
      float hv = fmaxf(ex * sw1[k] + ey * sw1[32 + k] + ez * sw1[64 + k] + sb1[k], 0.f);
      l0 += hv * sw2[k * 2];
      l1 += hv * sw2[k * 2 + 1];
    }
    row0[j] = nb ? l0 : FLT_MIN_REF;
    row1[j] = nb ? l1 : FLT_MIN_REF;
  }
  __syncthreads();

  for (int hsel = 0; hsel < 2; ++hsel) {
    float* row = hsel ? row1 : row0;
    float m = FLT_MIN_REF;
    for (int j = tid; j < NN; j += 256) m = fmaxf(m, row[j]);
    m = block_max(m, red);
    float s = 0.f;
    for (int j = tid; j < NN; j += 256) s += expf(row[j] - m);
    s = block_sum(s, red);
    float invs = 1.f / s;
    half_t* out = attn + ((size_t)(batch * 2 + hsel) * NN + i) * NN;
    for (int j = tid; j < NN; j += 256) out[j] = (half_t)(expf(row[j] - m) * invs);
    __syncthreads();
  }
}

// ---------------------------------------------------------------------------
// attn @ V via WMMA: per (b,h), (1024x1024 f16) @ (1024x64 f16) -> f16 out.
// A rows are contiguous 16B loads; B comes pre-packed (32B/lane contiguous).
// ---------------------------------------------------------------------------
__global__ void attnv_wmma(const half_t* __restrict__ attn, const half_t* __restrict__ vPack,
                           half_t* __restrict__ outH) {
  const int lane = threadIdx.x & 31;
  const int wave = threadIdx.x >> 5;
  const int w = blockIdx.x * 4 + wave;   // 4096 waves total
  const int bh = w >> 8;                 // 16 (b,h) pairs, 256 tiles each
  const int t  = w & 255;
  const int tm = t >> 2;                 // 64 row tiles
  const int tn = t & 3;                  // 4 col tiles (64 = 4*16)
  const int batch = bh >> 1, h = bh & 1;

  const half_t* A  = attn + (size_t)bh * NN * NN + (size_t)(tm * 16) * NN;
  const half_t* Vp = vPack + (size_t)bh * (32 * 4 * 512);

  const int mA    = lane & 15;
  const int kAoff = (lane < 16) ? 0 : 8;

  f32x8 acc = {};
#pragma unroll 4
  for (int kb = 0; kb < NN; kb += 32) {
    const half_t* ap = A + (size_t)mA * NN + kb + kAoff;
    if (kb + 32 < NN) __builtin_prefetch(ap + 32, 0, 0);   // next K-block
    half16 a;
    half8v alo = *(const half8v*)(ap);
    half8v ahi = *(const half8v*)(ap + 16);
#pragma unroll
    for (int i = 0; i < 8; ++i) { a[i] = alo[i]; a[8 + i] = ahi[i]; }
    const half_t* bp = Vp + ((size_t)((kb >> 5) * 4 + tn) * 32 + lane) * 16;
    half16 b = *(const half16*)bp;
    acc = __builtin_amdgcn_wmma_f32_16x16x32_f16(false, a, false, b,
                                                 (short)0, acc, false, false);
  }
  half_t* O = outH + (size_t)batch * NN * DD + h * 64;
  const int nB = tn * 16 + (lane & 15);
#pragma unroll
  for (int r = 0; r < 8; ++r) {
    int row = tm * 16 + ((lane < 16) ? r : (8 + r));
    O[(size_t)row * DD + nB] = (half_t)acc[r];
  }
}

// ---------------------------------------------------------------------------
// LayerNorm(128) + exact GELU -> latent_vecs (written straight into d_out)
// ---------------------------------------------------------------------------
__global__ void ln_gelu_kernel(const float* __restrict__ x, const float* __restrict__ g,
                               const float* __restrict__ b, float* __restrict__ out) {
  int t = blockIdx.x * blockDim.x + threadIdx.x;
  if (t >= TOK) return;
  const float* xr = x + (size_t)t * DD;
  float mu = 0.f;
  for (int c = 0; c < DD; ++c) mu += xr[c];
  mu *= (1.f / DD);
  float var = 0.f;
  for (int c = 0; c < DD; ++c) { float d = xr[c] - mu; var += d * d; }
  var *= (1.f / DD);
  float inv = rsqrtf(var + 1e-5f);
  for (int c = 0; c < DD; ++c) {
    float y = (xr[c] - mu) * inv * g[c] + b[c];
    out[(size_t)t * DD + c] = 0.5f * y * (1.f + erff(y * 0.70710678118654752f));
  }
}

// ---------------------------------------------------------------------------
// Heads: anchor head (sigmoid 4-way) + reg head + argmax + gather + add.
// ---------------------------------------------------------------------------
__global__ void heads_kernel(const float* __restrict__ latent,
                             const float* __restrict__ aw1, const float* __restrict__ ab1,
                             const float* __restrict__ aw2, const float* __restrict__ ab2,
                             const float* __restrict__ rw1, const float* __restrict__ rb1,
                             const float* __restrict__ rw2, const float* __restrict__ rb2,
                             const float* __restrict__ anchors,
                             float* __restrict__ match, float* __restrict__ depth,
                             float* __restrict__ idxout) {
  int t = blockIdx.x * blockDim.x + threadIdx.x;
  if (t >= TOK) return;
  const float* x = latent + (size_t)t * DD;
  float a[64];
#pragma unroll
  for (int j = 0; j < 64; ++j) a[j] = ab1[j];
  for (int c = 0; c < DD; ++c) {
    float xc = x[c];
#pragma unroll
    for (int j = 0; j < 64; ++j) a[j] += xc * aw1[c * 64 + j];
  }
#pragma unroll
  for (int j = 0; j < 64; ++j) a[j] = fmaxf(a[j], 0.f);
  float best = FLT_MIN_REF; int bi = 0;
#pragma unroll
  for (int k = 0; k < 4; ++k) {
    float s = ab2[k];
    for (int j = 0; j < 64; ++j) s += a[j] * aw2[j * 4 + k];
    float sg = 1.f / (1.f + expf(-s));
    match[(size_t)t * 4 + k] = sg;
    if (sg > best) { best = sg; bi = k; }   // first-max wins (strict >)
  }
#pragma unroll
  for (int j = 0; j < 64; ++j) a[j] = rb1[j];
  for (int c = 0; c < DD; ++c) {
    float xc = x[c];
#pragma unroll
    for (int j = 0; j < 64; ++j) a[j] += xc * rw1[c * 64 + j];
  }
  float d = rb2[0];
#pragma unroll
  for (int j = 0; j < 64; ++j) d += fmaxf(a[j], 0.f) * rw2[j];
  depth[t]  = anchors[bi] + d;
  idxout[t] = (float)bi;
}

// ---------------------------------------------------------------------------
extern "C" void kernel_launch(void* const* d_in, const int* in_sizes, int n_in,
                              void* d_out, int out_size, void* d_ws, size_t ws_size,
                              hipStream_t stream) {
  // setup_inputs() insertion order, params dict in init_params() insertion order
  const float* rays    = (const float*)d_in[0];
  const float* images  = (const float*)d_in[1];
  const float* coords  = (const float*)d_in[2];
  const float* anchors = (const float*)d_in[3];
  const float* geom_w1 = (const float*)d_in[4];
  const float* geom_b1 = (const float*)d_in[5];
  const float* geom_w2 = (const float*)d_in[6];
  const float* geom_b2 = (const float*)d_in[7];
  const float* geom_lg = (const float*)d_in[8];
  const float* geom_lb = (const float*)d_in[9];
  const float* conv1_w = (const float*)d_in[10];
  const float* conv1_b = (const float*)d_in[11];
  const float* conv2_w = (const float*)d_in[12];
  const float* conv2_b = (const float*)d_in[13];
  const float* conv3_w = (const float*)d_in[14];
  const float* conv3_b = (const float*)d_in[15];
  const float* rd_w1   = (const float*)d_in[16];
  const float* rd_b1   = (const float*)d_in[17];
  const float* rd_w2   = (const float*)d_in[18];
  const float* rd_b2   = (const float*)d_in[19];
  const float* v_w     = (const float*)d_in[20];
  const float* v_b     = (const float*)d_in[21];
  const float* o_w     = (const float*)d_in[22];
  const float* o_b     = (const float*)d_in[23];
  const float* d_w     = (const float*)d_in[24];
  const float* d_b     = (const float*)d_in[25];
  const float* d_lg    = (const float*)d_in[26];
  const float* d_lb    = (const float*)d_in[27];
  const float* anc_w1  = (const float*)d_in[28];
  const float* anc_b1  = (const float*)d_in[29];
  const float* anc_w2  = (const float*)d_in[30];
  const float* anc_b2  = (const float*)d_in[31];
  const float* reg_w1  = (const float*)d_in[32];
  const float* reg_b1  = (const float*)d_in[33];
  const float* reg_w2  = (const float*)d_in[34];
  const float* reg_b2  = (const float*)d_in[35];
  (void)in_sizes; (void)n_in; (void)out_size; (void)ws_size;

  // workspace carve-up (256B aligned)
  char* wp = (char*)d_ws;
  auto alloc = [&](size_t bytes) {
    void* p = (void*)wp;
    wp += (bytes + 255) & ~(size_t)255;
    return p;
  };
  float*  geom   = (float*) alloc((size_t)TOK * 32 * 4);
  float*  f1     = (float*) alloc((size_t)BB * 16 * 112 * 112 * 4);
  float*  f2     = (float*) alloc((size_t)BB * 32 * 56 * 56 * 4);
  float*  appear = (float*) alloc((size_t)BB * 96 * 4);
  float*  joint  = (float*) alloc((size_t)TOK * DD * 4);
  half_t* jointH = (half_t*)alloc((size_t)TOK * DD * 2);
  half_t* wvP    = (half_t*)alloc((size_t)DD * DD * 2);
  half_t* woP    = (half_t*)alloc((size_t)DD * DD * 2);
  half_t* wdP    = (half_t*)alloc((size_t)DD * DD * 2);
  half_t* vPack  = (half_t*)alloc((size_t)BB * 2 * 32 * 4 * 512 * 2);  // 2 MB
  half_t* attn   = (half_t*)alloc((size_t)BB * 2 * NN * NN * 2);       // 32 MB
  half_t* aoutH  = (half_t*)alloc((size_t)TOK * DD * 2);
  half_t* ooutH  = (half_t*)alloc((size_t)TOK * DD * 2);
  float*  dpre   = (float*) alloc((size_t)TOK * DD * 4);

  float* out_latent = (float*)d_out;                          // 8*1024*128
  float* out_match  = out_latent + (size_t)TOK * DD;          // 8*1024*4
  float* out_depth  = out_match + (size_t)TOK * 4;            // 8*1024
  float* out_idx    = out_depth + (size_t)TOK;                // 8*1024

  // 1. geometry branch
  geom_kernel<<<TOK / 256, 256, 0, stream>>>(rays, geom_w1, geom_b1, geom_w2, geom_b2,
                                             geom_lg, geom_lb, geom);
  // 2. appearance branch
  conv1_kernel<<<(BB * 16 * 112 * 112 + 255) / 256, 256, 0, stream>>>(images, conv1_w, conv1_b, f1);
  conv2_kernel<<<(BB * 32 * 56 * 56 + 255) / 256, 256, 0, stream>>>(f1, conv2_w, conv2_b, f2);
  conv3_mean_kernel<<<BB * 96, 256, 0, stream>>>(f2, conv3_w, conv3_b, appear);
  // 3. joint features + packed f16 weights
  joint_kernel<<<(TOK * DD) / 256, 256, 0, stream>>>(geom, appear, joint, jointH);
  packw_kernel<<<(DD * DD) / 256, 256, 0, stream>>>(v_w, o_w, d_w, wvP, woP, wdP);
  // 4. V projection (WMMA) -> directly into attn@V B-fragment layout
  gemm_wmma_128<<<(TOK / 16) * 8 / 4, 128, 0, stream>>>(jointH, wvP, v_b, nullptr,
                                                        nullptr, nullptr, vPack);
  // 5. fused logits + mask + softmax -> f16 attention
  attn_softmax_kernel<<<BB * NN, 256, 0, stream>>>(rays, coords, rd_w1, rd_b1, rd_w2, rd_b2, attn);
  // 6. attn @ V (WMMA)
  attnv_wmma<<<(BB * 2 * 64 * 4) / 4, 128, 0, stream>>>(attn, vPack, aoutH);
  // 7. output projection + residual (WMMA)
  gemm_wmma_128<<<(TOK / 16) * 8 / 4, 128, 0, stream>>>(aoutH, woP, o_b, joint,
                                                        nullptr, ooutH, nullptr);
  // 8. decoder projection (WMMA)
  gemm_wmma_128<<<(TOK / 16) * 8 / 4, 128, 0, stream>>>(ooutH, wdP, d_b, nullptr,
                                                        dpre, nullptr, nullptr);
  // 9. LN + GELU -> latent
  ln_gelu_kernel<<<TOK / 256, 256, 0, stream>>>(dpre, d_lg, d_lb, out_latent);
  // 10. heads
  heads_kernel<<<TOK / 256, 256, 0, stream>>>(out_latent, anc_w1, anc_b1, anc_w2, anc_b2,
                                              reg_w1, reg_b1, reg_w2, reg_b2, anchors,
                                              out_match, out_depth, out_idx);
}
